// TransformerLayer_54262616818144
// MI455X (gfx1250) — compile-verified, matrix operand order checked by reading, not run on hardware
//
#include <hip/hip_runtime.h>

// ---------------- Problem constants (from setup_inputs) ----------------
#define SEQ   2048
#define BATCH 2
#define HDIM  1024
#define NHEAD 16
#define NKV   4
#define HDHD  64
#define QKV_N 1536      // (NH + 2*NKV) * HD
#define NEXP  8
#define FDIM  4096
#define TOKS  (SEQ * BATCH)   // 4096
#define NASSIGN (TOKS * 2)    // 8192 (top_k = 2)

typedef __attribute__((ext_vector_type(16))) __bf16 v16bf;
typedef __attribute__((ext_vector_type(8)))  float  v8f;

// ---------------- WMMA helpers ----------------
__device__ __forceinline__ v8f wmma_bf16(v16bf a, v16bf b, v8f c) {
  return __builtin_amdgcn_wmma_f32_16x16x32_bf16(false, a, false, b, (short)0, c,
                                                 false, false);
}

// A-fragment style access: lane = row (m), elements sweep K per ISA layout.
// K indices for lane half h: {h*8+e, 16+h*8+e}
__device__ __forceinline__ v16bf frag_row(const __bf16* p, int ldm, int lane,
                                          int colOff) {
  int m = lane & 15, h = lane >> 4;
  const __bf16* r = p + (size_t)m * ldm + colOff + h * 8;
  v16bf f;
#pragma unroll
  for (int e = 0; e < 8; ++e) f[e] = r[e];
#pragma unroll
  for (int e = 0; e < 8; ++e) f[e + 8] = r[16 + e];
  return f;
}

// B-fragment from a row-major [K][N] tile: lane = column (n), elements sweep K.
__device__ __forceinline__ v16bf frag_col(const __bf16* p, int ldm, int lane,
                                          int colOff) {
  int n = lane & 15, h = lane >> 4;
  const __bf16* r = p + (size_t)(h * 8) * ldm + colOff + n;
  v16bf f;
#pragma unroll
  for (int e = 0; e < 8; ++e) f[e] = r[(size_t)e * ldm];
#pragma unroll
  for (int e = 0; e < 8; ++e) f[e + 8] = r[(size_t)(16 + e) * ldm];
  return f;
}

// ---------------- CDNA5 async global->LDS staging ----------------
// 16B per lane, tracked by ASYNCcnt. dst must be a wave-relative LDS byte
// address (low 32 bits of the flat shared pointer), 16B aligned.
__device__ __forceinline__ void async_copy_b128(void* lds_dst,
                                                const void* gsrc) {
  unsigned d = (unsigned)(uintptr_t)lds_dst;
  unsigned long long s = (unsigned long long)(uintptr_t)gsrc;
  asm volatile("global_load_async_to_lds_b128 %0, %1, off"
               :
               : "v"(d), "v"(s)
               : "memory");
}
__device__ __forceinline__ void async_wait0() {
  asm volatile("s_wait_asynccnt 0" ::: "memory");
}
// allow the newest 4 transfers (the prefetched tile) to remain in flight
__device__ __forceinline__ void async_wait4() {
  asm volatile("s_wait_asynccnt 4" ::: "memory");
}

// ---------------- DPP16 butterfly reductions (per 16-lane half) -----------
// xor1, xor2, xor7(row_half_mirror), xor15(row_mirror) pairings: every lane
// of a 16-lane half ends with the reduction of that half. Pure VALU, no DS.
__device__ __forceinline__ float dpp_max16(float x) {
  int t;
  t = __builtin_amdgcn_update_dpp(0, __builtin_bit_cast(int, x), 0xB1, 0xF,
                                  0xF, true);  // quad_perm [1,0,3,2]
  x = fmaxf(x, __builtin_bit_cast(float, t));
  t = __builtin_amdgcn_update_dpp(0, __builtin_bit_cast(int, x), 0x4E, 0xF,
                                  0xF, true);  // quad_perm [2,3,0,1]
  x = fmaxf(x, __builtin_bit_cast(float, t));
  t = __builtin_amdgcn_update_dpp(0, __builtin_bit_cast(int, x), 0x141, 0xF,
                                  0xF, true);  // row_half_mirror
  x = fmaxf(x, __builtin_bit_cast(float, t));
  t = __builtin_amdgcn_update_dpp(0, __builtin_bit_cast(int, x), 0x140, 0xF,
                                  0xF, true);  // row_mirror
  x = fmaxf(x, __builtin_bit_cast(float, t));
  return x;
}
__device__ __forceinline__ float dpp_sum16(float x) {
  int t;
  t = __builtin_amdgcn_update_dpp(0, __builtin_bit_cast(int, x), 0xB1, 0xF,
                                  0xF, true);
  x += __builtin_bit_cast(float, t);
  t = __builtin_amdgcn_update_dpp(0, __builtin_bit_cast(int, x), 0x4E, 0xF,
                                  0xF, true);
  x += __builtin_bit_cast(float, t);
  t = __builtin_amdgcn_update_dpp(0, __builtin_bit_cast(int, x), 0x141, 0xF,
                                  0xF, true);
  x += __builtin_bit_cast(float, t);
  t = __builtin_amdgcn_update_dpp(0, __builtin_bit_cast(int, x), 0x140, 0xF,
                                  0xF, true);
  x += __builtin_bit_cast(float, t);
  return x;
}

// ---------------- fp32 -> bf16 conversion ----------------
__global__ __launch_bounds__(256) void f2bf_kernel(const float* __restrict__ in,
                                                   __bf16* __restrict__ out,
                                                   long n) {
  long i = (long)blockIdx.x * 256 + threadIdx.x;
  long stride = (long)gridDim.x * 256;
  for (; i < n; i += stride) out[i] = (__bf16)in[i];
}

// ---------------- LayerNorm -> bf16 ----------------
__global__ __launch_bounds__(256) void ln_bf16_kernel(
    const float* __restrict__ x, const float* __restrict__ w,
    const float* __restrict__ bia, __bf16* __restrict__ out) {
  int t = blockIdx.x;
  const float* xr = x + (size_t)t * HDIM;
  float s = 0.f, s2 = 0.f;
  for (int h = threadIdx.x; h < HDIM; h += 256) {
    float v = xr[h];
    s += v;
    s2 += v * v;
  }
  __shared__ float r1[256], r2[256];
  r1[threadIdx.x] = s;
  r2[threadIdx.x] = s2;
  __syncthreads();
  for (int o = 128; o > 0; o >>= 1) {
    if (threadIdx.x < o) {
      r1[threadIdx.x] += r1[threadIdx.x + o];
      r2[threadIdx.x] += r2[threadIdx.x + o];
    }
    __syncthreads();
  }
  float mu = r1[0] * (1.0f / HDIM);
  float var = r2[0] * (1.0f / HDIM) - mu * mu;
  float rs = rsqrtf(var + 1e-5f);
  for (int h = threadIdx.x; h < HDIM; h += 256)
    out[(size_t)t * HDIM + h] = (__bf16)((xr[h] - mu) * rs * w[h] + bia[h]);
}

// ---------------- Generic WMMA GEMM tile body ----------------
// C[128x128] tile at (m0, n0). 256 threads = 8 waves in a 4x2 grid; wave
// (wr,wc) owns rows wr*32..+31, cols wc*64..+63 -> 8 WMMAs per K-step.
// Double-buffered LDS with ASYNCcnt-throttled prefetch (4 B128 ops/thread
// per tile -> s_wait_asynccnt 4 keeps the next tile streaming in).
// EPI: 0 = bf16 out, 1 = f32 out + residual, 2 = GELU -> bf16, 3 = f32 out.
template <int EPI>
__device__ __forceinline__ void gemm_tile_body(
    const __bf16* __restrict__ A, int lda, const int* __restrict__ rowmap,
    int m0, int Mv, const __bf16* __restrict__ B, int N, int K, int n0,
    __bf16* __restrict__ outB, float* __restrict__ outF, int ldo,
    const float* __restrict__ res) {
  __shared__ __align__(16) __bf16 As[2][128][40];  // 80B rows (16B mult)
  __shared__ __align__(16) __bf16 Bs[2][32][136];  // 272B rows (16B mult)
  __shared__ int rows_s[128];
  int tid = threadIdx.x;
  if (tid < 128) {
    int r = m0 + tid;
    // clamp invalid rows to 0: their garbage only feeds C rows masked in the
    // epilogue, and row 0 is always a valid address.
    rows_s[tid] = (r < Mv) ? (rowmap ? rowmap[r] : r) : 0;
  }
  __syncthreads();
  int w = tid >> 5, lane = tid & 31;
  int wr = w >> 1, wc = w & 1;

  auto stage = [&](int buf, int k0) {
    // A tile: 128x32 = 512 x 16B transfers (2/thread)
    for (int x = tid; x < 512; x += 256) {
      int i = x >> 2, kk8 = (x & 3) * 8;
      async_copy_b128(&As[buf][i][kk8], A + (size_t)rows_s[i] * lda + k0 + kk8);
    }
    // B tile: 32x128 = 512 x 16B transfers (2/thread)
    for (int x = tid; x < 512; x += 256) {
      int kk = x >> 4, j8 = (x & 15) * 8;
      async_copy_b128(&Bs[buf][kk][j8], B + (size_t)(k0 + kk) * N + n0 + j8);
    }
  };

  v8f acc[2][4] = {};
  int KT = K / 32;
  stage(0, 0);
  for (int kt = 0; kt < KT; ++kt) {
    int cur = kt & 1;
    if (kt + 1 < KT) {
      stage(cur ^ 1, (kt + 1) * 32);
      async_wait4();  // tile kt resident (in-order completion), kt+1 in flight
    } else {
      async_wait0();
    }
    __syncthreads();
    v16bf a0 = frag_row(&As[cur][wr * 32][0], 40, lane, 0);
    v16bf a1 = frag_row(&As[cur][wr * 32 + 16][0], 40, lane, 0);
#pragma unroll
    for (int j = 0; j < 4; ++j) {
      v16bf bf = frag_col(&Bs[cur][0][0], 136, lane, wc * 64 + j * 16);
      acc[0][j] = wmma_bf16(a0, bf, acc[0][j]);
      acc[1][j] = wmma_bf16(a1, bf, acc[1][j]);
    }
    __syncthreads();  // all waves done reading buf before it is restaged
  }
  int n = lane & 15, half = lane >> 4;
#pragma unroll
  for (int ai = 0; ai < 2; ++ai) {
#pragma unroll
    for (int j = 0; j < 4; ++j) {
#pragma unroll
      for (int r = 0; r < 8; ++r) {
        int row = m0 + wr * 32 + ai * 16 + 8 * half + r;
        if (row >= Mv) continue;
        int col = n0 + wc * 64 + j * 16 + n;
        float v = acc[ai][j][r];
        size_t o = (size_t)row * ldo + col;
        if (EPI == 0) {
          outB[o] = (__bf16)v;
        } else if (EPI == 1) {
          outF[o] = v + res[o];
        } else if (EPI == 2) {
          float g = 0.5f * v * (1.0f + erff(v * 0.70710678118f));
          outB[o] = (__bf16)g;
        } else {
          outF[o] = v;
        }
      }
    }
  }
}

template <int EPI>
__global__ __launch_bounds__(256) void wmma_gemm_dense(
    const __bf16* __restrict__ A, int lda, const __bf16* __restrict__ B, int N,
    int K, __bf16* __restrict__ outB, float* __restrict__ outF, int ldo,
    const float* __restrict__ res, int M) {
  gemm_tile_body<EPI>(A, lda, nullptr, blockIdx.y * 128, M, B, N, K,
                      blockIdx.x * 128, outB, outF, ldo, res);
}

template <int EPI>
__global__ __launch_bounds__(256) void wmma_gemm_moe(
    const __bf16* __restrict__ A, int lda, const int* __restrict__ rowmap,
    const __bf16* __restrict__ Ball, long bstride, int N, int K,
    __bf16* __restrict__ outB, float* __restrict__ outF, int ldo,
    const int* __restrict__ cnt, const int* __restrict__ off) {
  int e = blockIdx.z;
  int c = cnt[e];
  if ((int)blockIdx.y * 128 >= c) return;  // uniform per block
  int m0 = off[e] + blockIdx.y * 128;
  int Mv = off[e] + c;
  gemm_tile_body<EPI>(A, lda, rowmap, m0, Mv, Ball + (size_t)e * bstride, N, K,
                      blockIdx.x * 128, outB, outF, ldo, nullptr);
}

// ---------------- Flash attention (causal, GQA) ----------------
// grid: (B*NH, SEQ/64). block: 128 threads = 4 waves, each wave owns 16 q rows.
__global__ __launch_bounds__(128) void attn_kernel(
    const __bf16* __restrict__ qkvb, __bf16* __restrict__ attnb) {
  __shared__ __align__(16) __bf16 Ks[32][72];  // 144B rows: 16B mult
  __shared__ __align__(16) __bf16 Vs[32][72];
  __shared__ __bf16 Ps[4][16][34];
  int bh = blockIdx.x;
  int b = bh / NHEAD, h = bh % NHEAD;
  int hk = h >> 2;  // NH/NKV = 4
  int q0 = blockIdx.y * 64;
  int tid = threadIdx.x;
  int w = tid >> 5, lane = tid & 31;
  int nIdx = lane & 15, half = lane >> 4;

  // Q fragments for this wave's 16 rows (d = 0..31 and 32..63)
  const __bf16* qbase =
      qkvb + ((size_t)(q0 + w * 16) * BATCH + b) * QKV_N + h * HDHD;
  v16bf qa0 = frag_row(qbase, BATCH * QKV_N, lane, 0);
  v16bf qa1 = frag_row(qbase, BATCH * QKV_N, lane, 32);

  float m_run[8], l_run[8];
#pragma unroll
  for (int r = 0; r < 8; ++r) {
    m_run[r] = -3.0e38f;
    l_run[r] = 0.0f;
  }
  v8f acc[4] = {};
  const float scale = 0.125f;  // 1/sqrt(64)

  for (int j0 = 0; j0 < q0 + 64; j0 += 32) {
    // stage K and V blocks (32 keys x 64 dims each) via async DMA:
    // 2 * 32 * 8 = 512 x 16B transfers, 4 per thread.
    for (int x = tid; x < 512; x += 128) {
      int isV = x >> 8;
      int key = (x >> 3) & 31;
      int d8 = (x & 7) * 8;
      const __bf16* src = qkvb + ((size_t)(j0 + key) * BATCH + b) * QKV_N +
                          (isV ? 1280 : 1024) + hk * HDHD + d8;
      async_copy_b128(isV ? &Vs[key][d8] : &Ks[key][d8], src);
    }
    async_wait0();
    __syncthreads();

    // scores: two 16x16 n-subtiles, chained over d halves
    v8f z = {};
    v8f s0 = wmma_bf16(qa1, frag_row(&Ks[0][0], 72, lane, 32),
                       wmma_bf16(qa0, frag_row(&Ks[0][0], 72, lane, 0), z));
    v8f s1 = wmma_bf16(qa1, frag_row(&Ks[16][0], 72, lane, 32),
                       wmma_bf16(qa0, frag_row(&Ks[16][0], 72, lane, 0), z));

    float corr[8], p0v[8], p1v[8];
#pragma unroll
    for (int r = 0; r < 8; ++r) {
      int qg = q0 + w * 16 + 8 * half + r;
      float a = (j0 + nIdx <= qg) ? s0[r] * scale : -3.0e38f;
      float c2 = (j0 + 16 + nIdx <= qg) ? s1[r] * scale : -3.0e38f;
      float mx = dpp_max16(fmaxf(a, c2));
      float mn = fmaxf(m_run[r], mx);
      corr[r] = __expf(m_run[r] - mn);
      float e0 = __expf(a - mn), e1 = __expf(c2 - mn);
      float sum = dpp_sum16(e0 + e1);
      l_run[r] = l_run[r] * corr[r] + sum;
      m_run[r] = mn;
      p0v[r] = e0;
      p1v[r] = e1;
    }
#pragma unroll
    for (int j = 0; j < 4; ++j)
#pragma unroll
      for (int r = 0; r < 8; ++r) acc[j][r] *= corr[r];

    // write P (16x32, bf16) for this wave
#pragma unroll
    for (int r = 0; r < 8; ++r) {
      Ps[w][8 * half + r][nIdx] = (__bf16)p0v[r];
      Ps[w][8 * half + r][16 + nIdx] = (__bf16)p1v[r];
    }
    __syncthreads();

    v16bf pa = frag_row(&Ps[w][0][0], 34, lane, 0);
#pragma unroll
    for (int jd = 0; jd < 4; ++jd) {
      v16bf vb = frag_col(&Vs[0][0], 72, lane, jd * 16);
      acc[jd] = wmma_bf16(pa, vb, acc[jd]);
    }
    __syncthreads();
  }

  // epilogue: normalize, write bf16 rows of attn_out [T][H]
#pragma unroll
  for (int jd = 0; jd < 4; ++jd) {
#pragma unroll
    for (int r = 0; r < 8; ++r) {
      int s = q0 + w * 16 + 8 * half + r;
      float o = acc[jd][r] / l_run[r];
      attnb[((size_t)s * BATCH + b) * HDIM + h * HDHD + jd * 16 + nIdx] =
          (__bf16)o;
    }
  }
}

// ---------------- Router / top-k / bucketing ----------------
__global__ __launch_bounds__(256) void router_kernel(
    const __bf16* __restrict__ x, const float* __restrict__ rw,
    float* __restrict__ logits) {
  int t = blockIdx.x;
  float p[NEXP] = {};
  for (int h = threadIdx.x; h < HDIM; h += 256) {
    float xv = (float)x[(size_t)t * HDIM + h];
#pragma unroll
    for (int e = 0; e < NEXP; ++e) p[e] += xv * rw[h * NEXP + e];
  }
  __shared__ float red[256];
  for (int e = 0; e < NEXP; ++e) {
    red[threadIdx.x] = p[e];
    __syncthreads();
    for (int o = 128; o > 0; o >>= 1) {
      if (threadIdx.x < o) red[threadIdx.x] += red[threadIdx.x + o];
      __syncthreads();
    }
    if (threadIdx.x == 0) logits[(size_t)t * NEXP + e] = red[0];
    __syncthreads();
  }
}

__global__ __launch_bounds__(256) void topk_kernel(
    const float* __restrict__ logits, float* __restrict__ top_p,
    int* __restrict__ top_i, int* __restrict__ counts) {
  int t = blockIdx.x * 256 + threadIdx.x;
  if (t >= TOKS) return;
  float l[NEXP];
  float mx = -3.0e38f;
#pragma unroll
  for (int e = 0; e < NEXP; ++e) {
    l[e] = logits[(size_t)t * NEXP + e];
    mx = fmaxf(mx, l[e]);
  }
  float den = 0.f;
#pragma unroll
  for (int e = 0; e < NEXP; ++e) {
    l[e] = __expf(l[e] - mx);
    den += l[e];
  }
  float rden = 1.0f / den;
#pragma unroll
  for (int e = 0; e < NEXP; ++e) l[e] *= rden;
  int i0 = 0;
#pragma unroll
  for (int e = 1; e < NEXP; ++e)
    if (l[e] > l[i0]) i0 = e;
  int i1 = (i0 == 0) ? 1 : 0;
#pragma unroll
  for (int e = 0; e < NEXP; ++e)
    if (e != i0 && l[e] > l[i1]) i1 = e;
  top_p[t * 2] = l[i0];
  top_p[t * 2 + 1] = l[i1];
  top_i[t * 2] = i0;
  top_i[t * 2 + 1] = i1;
  atomicAdd(&counts[i0], 1);
  atomicAdd(&counts[i1], 1);
}

__global__ void offsets_kernel(const int* __restrict__ counts,
                               int* __restrict__ off) {
  if (threadIdx.x == 0 && blockIdx.x == 0) {
    int a = 0;
    for (int e = 0; e < NEXP; ++e) {
      off[e] = a;
      a += counts[e];
    }
  }
}

__global__ __launch_bounds__(256) void scatter_kernel(
    const int* __restrict__ top_i, const int* __restrict__ off,
    int* __restrict__ cursor, int* __restrict__ assign_token,
    int* __restrict__ assign_of) {
  int t = blockIdx.x * 256 + threadIdx.x;
  if (t >= TOKS) return;
  for (int s = 0; s < 2; ++s) {
    int e = top_i[t * 2 + s];
    int pos = atomicAdd(&cursor[e], 1);
    int a = off[e] + pos;
    assign_token[a] = t;
    assign_of[t * 2 + s] = a;
  }
}

// ---------------- Final combine ----------------
__global__ __launch_bounds__(256) void combine_kernel(
    const float* __restrict__ h_attn, const float* __restrict__ ybuf,
    const float* __restrict__ top_p, const int* __restrict__ assign_of,
    float* __restrict__ out) {
  int t = blockIdx.x;
  int a0 = assign_of[t * 2], a1 = assign_of[t * 2 + 1];
  float p0 = top_p[t * 2], p1 = top_p[t * 2 + 1];
  for (int h = threadIdx.x; h < HDIM; h += 256) {
    size_t o = (size_t)t * HDIM + h;
    out[o] = h_attn[o] + p0 * ybuf[(size_t)a0 * HDIM + h] +
             p1 * ybuf[(size_t)a1 * HDIM + h];
  }
}

// ---------------- Host orchestration ----------------
extern "C" void kernel_launch(void* const* d_in, const int* in_sizes, int n_in,
                              void* d_out, int out_size, void* d_ws,
                              size_t ws_size, hipStream_t stream) {
  const float* hs = (const float*)d_in[0];
  const float* ln1w = (const float*)d_in[1];
  const float* ln1bia = (const float*)d_in[2];
  const float* ln2w = (const float*)d_in[3];
  const float* ln2bia = (const float*)d_in[4];
  const float* qkvw = (const float*)d_in[5];
  const float* projw = (const float*)d_in[6];
  const float* routerw = (const float*)d_in[7];
  const float* moew1 = (const float*)d_in[8];
  const float* moew2 = (const float*)d_in[9];
  float* out = (float*)d_out;

  char* ws = (char*)d_ws;
  size_t o = 0;
  auto alloc = [&](size_t bytes) {
    size_t r = o;
    o = (o + bytes + 255) & ~(size_t)255;
    return r;
  };
  __bf16* ln1b = (__bf16*)(ws + alloc((size_t)TOKS * HDIM * 2));
  __bf16* qkvwb = (__bf16*)(ws + alloc((size_t)HDIM * QKV_N * 2));
  __bf16* qkvb = (__bf16*)(ws + alloc((size_t)TOKS * QKV_N * 2));
  __bf16* attnb = (__bf16*)(ws + alloc((size_t)TOKS * HDIM * 2));
  __bf16* projb = (__bf16*)(ws + alloc((size_t)HDIM * HDIM * 2));
  float* hattn = (float*)(ws + alloc((size_t)TOKS * HDIM * 4));
  __bf16* ln2b = (__bf16*)(ws + alloc((size_t)TOKS * HDIM * 2));
  __bf16* w1b = (__bf16*)(ws + alloc((size_t)NEXP * HDIM * FDIM * 2));
  __bf16* w2b = (__bf16*)(ws + alloc((size_t)NEXP * FDIM * HDIM * 2));
  __bf16* hidden = (__bf16*)(ws + alloc((size_t)NASSIGN * FDIM * 2));
  float* ybuf = (float*)(ws + alloc((size_t)NASSIGN * HDIM * 4));
  float* logits = (float*)(ws + alloc((size_t)TOKS * NEXP * 4));
  float* top_p = (float*)(ws + alloc((size_t)TOKS * 2 * 4));
  int* top_i = (int*)(ws + alloc((size_t)TOKS * 2 * 4));
  int* counts = (int*)(ws + alloc(64));  // counts[8] + cursor[8]
  int* cursor = counts + 8;
  int* off = (int*)(ws + alloc(64));
  int* assign_token = (int*)(ws + alloc((size_t)NASSIGN * 4));
  int* assign_of = (int*)(ws + alloc((size_t)TOKS * 2 * 4));

  // 1. convert weights to bf16
  f2bf_kernel<<<512, 256, 0, stream>>>(qkvw, qkvwb, (long)HDIM * QKV_N);
  f2bf_kernel<<<512, 256, 0, stream>>>(projw, projb, (long)HDIM * HDIM);
  f2bf_kernel<<<2048, 256, 0, stream>>>(moew1, w1b, (long)NEXP * HDIM * FDIM);
  f2bf_kernel<<<2048, 256, 0, stream>>>(moew2, w2b, (long)NEXP * FDIM * HDIM);

  // 2. LN1
  ln_bf16_kernel<<<TOKS, 256, 0, stream>>>(hs, ln1w, ln1bia, ln1b);

  // 3. qkv = ln1 @ qkv_weight  (bf16 out)
  wmma_gemm_dense<0><<<dim3(QKV_N / 128, TOKS / 128), 256, 0, stream>>>(
      ln1b, HDIM, qkvwb, QKV_N, HDIM, qkvb, nullptr, QKV_N, nullptr, TOKS);

  // 4. causal GQA flash attention
  attn_kernel<<<dim3(BATCH * NHEAD, SEQ / 64), 128, 0, stream>>>(qkvb, attnb);

  // 5. h_attn = hidden + attn_out @ proj  (f32 out + residual)
  wmma_gemm_dense<1><<<dim3(HDIM / 128, TOKS / 128), 256, 0, stream>>>(
      attnb, HDIM, projb, HDIM, HDIM, nullptr, hattn, HDIM, hs, TOKS);

  // 6. LN2
  ln_bf16_kernel<<<TOKS, 256, 0, stream>>>(hattn, ln2w, ln2bia, ln2b);

  // 7. router logits, top-2, bucketize
  router_kernel<<<TOKS, 256, 0, stream>>>(ln2b, routerw, logits);
  hipMemsetAsync(counts, 0, 64, stream);
  topk_kernel<<<TOKS / 256, 256, 0, stream>>>(logits, top_p, top_i, counts);
  offsets_kernel<<<1, 32, 0, stream>>>(counts, off);
  scatter_kernel<<<TOKS / 256, 256, 0, stream>>>(top_i, off, cursor,
                                                 assign_token, assign_of);

  // 8. expert GEMM 1: hidden = gelu(gather(ln2) @ w1[e])  (bf16 out)
  wmma_gemm_moe<2><<<dim3(FDIM / 128, NASSIGN / 128, NEXP), 256, 0, stream>>>(
      ln2b, HDIM, assign_token, w1b, (long)HDIM * FDIM, FDIM, HDIM, hidden,
      nullptr, FDIM, counts, off);

  // 9. expert GEMM 2: y = hidden @ w2[e]  (f32 out)
  wmma_gemm_moe<3><<<dim3(HDIM / 128, NASSIGN / 128, NEXP), 256, 0, stream>>>(
      hidden, FDIM, nullptr, w2b, (long)FDIM * HDIM, HDIM, FDIM, nullptr, ybuf,
      HDIM, counts, off);

  // 10. out = h_attn + sum_k p_k * y_k
  combine_kernel<<<TOKS, 256, 0, stream>>>(hattn, ybuf, top_p, assign_of, out);
  (void)in_sizes;
  (void)n_in;
  (void)out_size;
  (void)ws_size;
}
